// MiniTransformerBlock_31155692765526
// MI455X (gfx1250) — compile-verified, matrix operand order checked by reading, not run on hardware
//
#include <hip/hip_runtime.h>
#include <hip/hip_bf16.h>

// ---------------------------------------------------------------------------
// MiniTransformerBlock on gfx1250 (MI455X): bf16 WMMA + TDM (tensor DMA).
//   h   = RMSNorm(emb[x]) * norm_w              (f32 -> bf16)
//   Q   = h @ Wq^T,  K = h @ Wk^T               (bf16 WMMA, f32 acc -> bf16)
//   V^T = (h @ Wv^T)^T                          (stored transposed for PV GEMM)
//   L   = Q @ K^T                               (f32 logits, lives in 192MB L2)
//   P   = softmax_rows(L)                       (bf16)
//   out = silu(P @ V)                           (f32)
// GEMM: 64x256 block tile; K-slices staged into double-buffered LDS by the
// Tensor Data Mover (tensor_load_to_lds, TENSORcnt) issued by wave 0 only.
// Waves (2Mx4N) each compute 32x64 via v_wmma_f32_16x16x32_bf16 fed by
// ds_load_b128 from the shared tiles.
// ---------------------------------------------------------------------------

#define SEQ   4096
#define DIM   1024

typedef __attribute__((ext_vector_type(8)))  unsigned short ushort8;
typedef __attribute__((ext_vector_type(16))) unsigned short ushort16;
typedef __attribute__((ext_vector_type(16))) __bf16         bf16x16;
typedef __attribute__((ext_vector_type(8)))  float          v8f;
typedef __attribute__((ext_vector_type(4)))  unsigned int   uint32x4;
typedef __attribute__((ext_vector_type(8)))  int            int32x8;
typedef __attribute__((ext_vector_type(4)))  int            int32x4;

// ---- f32 -> bf16 (round to nearest even) ----------------------------------
__device__ __forceinline__ unsigned short f2bf(float f) {
  unsigned u = __builtin_bit_cast(unsigned, f);
  u += 0x7FFFu + ((u >> 16) & 1u);
  return (unsigned short)(u >> 16);
}

// ---- CDNA5 Tensor Data Mover: 2D bf16 tile, global -> LDS -----------------
// D# per ISA 8.3/8.4: count=1, type=2 ("image"), data_size=2B.
// pad_enable with interval code 3 (16 dwords = one 32-elem row) and amount
// code 3 (4 dwords = 16B) yields an 80B LDS row pitch = our [..][40] arrays.
// tensor dims == tile dims (tile never goes OOB); dim0 stride = row pitch.
// This toolchain's builtin is the 6-arg form:
//   (uint32x4 g0, int32x8 g1, int32x4 g2, int32x4 g3, int32x8, i32 cpol)
__device__ __forceinline__ void tdm_load_2d(unsigned ldsAddr, const void* gaddr,
                                            unsigned strideD0,   // elements
                                            unsigned tileD0,     // elements
                                            unsigned tileD1) {   // rows
  unsigned long long ga = (unsigned long long)(uintptr_t)gaddr;
  uint32x4 g0 = { 1u,                                   // count=1, user desc
                  ldsAddr,                              // lds_addr (bytes)
                  (unsigned)ga,                         // global_addr[31:0]
                  ((unsigned)(ga >> 32) & 0x01FFFFFFu)  // global_addr[56:32]
                      | 0x80000000u };                  // type=2
  int32x8 g1 = {
      (int)((1u << 16)        // data_size = 2 bytes
          | (1u << 20)        // pad_enable
          | (3u << 22)        // pad_interval: 16 dwords
          | (3u << 25)),      // pad_amount : 4 dwords
      (int)((tileD0 & 0xFFFFu) << 16),                            // tensor_dim0 lo
      (int)(((tileD0 >> 16) & 0xFFFFu) | ((tileD1 & 0xFFFFu) << 16)), // d0 hi|d1 lo
      (int)(((tileD1 >> 16) & 0xFFFFu) | ((tileD0 & 0xFFFFu) << 16)), // d1 hi|tile_dim0
      (int)(tileD1 & 0xFFFFu),                                    // tile_dim1|tile_dim2=0
      (int)strideD0,                                              // tensor_dim0_stride lo32
      0, 0 };                                                     // stride hi, dim1_stride
  int32x4 gz4 = { 0, 0, 0, 0 };                                   // 2D: groups 2/3 unused
  int32x8 gz8 = { 0, 0, 0, 0, 0, 0, 0, 0 };
  __builtin_amdgcn_tensor_load_to_lds(g0, g1, gz4, gz4, gz8, 0);
}

// ---- build a WMMA bf16 operand from two 16B chunks ------------------------
// Striping per ISA 7.12.2 (16-bit A/B 16x32):
//   lanes 0-15 : row = lane,    K = {0..7} and {16..23}
//   lanes 16-31: row = lane-16, K = {8..15} and {24..31}
__device__ __forceinline__ bf16x16 make_op(const unsigned short* p) {
  ushort8 lo = *(const ushort8*)p;
  ushort8 hi = *(const ushort8*)(p + 16);
  ushort16 v = __builtin_shufflevector(lo, hi, 0,1,2,3,4,5,6,7,8,9,10,11,12,13,14,15);
  return __builtin_bit_cast(bf16x16, v);
}

// ---------------------------------------------------------------------------
// Kernel 1: embedding gather + RMSNorm -> bf16 h
// ---------------------------------------------------------------------------
__global__ __launch_bounds__(256) void embed_rmsnorm_kernel(
    const int* __restrict__ x, const float* __restrict__ emb,
    const float* __restrict__ nw, unsigned short* __restrict__ hb) {
  int tok_pos = blockIdx.x;
  int tok     = x[tok_pos];
  const float4* e4 = reinterpret_cast<const float4*>(emb + (size_t)tok * DIM);
  float4 v = e4[threadIdx.x];                         // 256 threads * 4 = 1024
  float ss = v.x * v.x + v.y * v.y + v.z * v.z + v.w * v.w;

  for (int o = 16; o > 0; o >>= 1) ss += __shfl_xor(ss, o, 32);
  __shared__ float red[8];
  int wave = threadIdx.x >> 5, lane = threadIdx.x & 31;
  if (lane == 0) red[wave] = ss;
  __syncthreads();
  float tot = 0.f;
#pragma unroll
  for (int i = 0; i < 8; ++i) tot += red[i];
  float scale = rsqrtf(tot * (1.0f / DIM) + 1.1920929e-07f);  // eps = FLT_EPSILON

  const float4* w4 = reinterpret_cast<const float4*>(nw);
  float4 w = w4[threadIdx.x];
  size_t base = (size_t)tok_pos * DIM + threadIdx.x * 4;
  hb[base + 0] = f2bf(v.x * scale * w.x);
  hb[base + 1] = f2bf(v.y * scale * w.y);
  hb[base + 2] = f2bf(v.z * scale * w.z);
  hb[base + 3] = f2bf(v.w * scale * w.w);
}

// ---------------------------------------------------------------------------
// Kernel 2: f32 -> bf16 bulk convert (weights)
// ---------------------------------------------------------------------------
__global__ __launch_bounds__(256) void cvt_f32_bf16_kernel(
    const float* __restrict__ in, unsigned short* __restrict__ out, int n4) {
  int i = blockIdx.x * 256 + threadIdx.x;
  if (i < n4) {
    float4 v = reinterpret_cast<const float4*>(in)[i];
    out[i * 4 + 0] = f2bf(v.x);
    out[i * 4 + 1] = f2bf(v.y);
    out[i * 4 + 2] = f2bf(v.z);
    out[i * 4 + 3] = f2bf(v.w);
  }
}

// ---------------------------------------------------------------------------
// Kernel 3: WMMA GEMM  C[M][N] = A[M][K] @ Bt[N][K]^T
//   Block = 8 waves as 2(M) x 4(N): block tile 64 x 256.
//   Wave computes 32 x 64 (2 A-tiles x 4 B-tiles = 8 WMMA / K-step).
//   K staged in 32-deep slices into double-buffered LDS by the TDM
//   (wave 0 issues tensor_load_to_lds; TENSORcnt + barrier per K-step).
//   MODE 0: store bf16 row-major   MODE 1: store bf16 transposed ([N][M])
//   MODE 2: store f32 + SiLU       MODE 3: store f32
// ---------------------------------------------------------------------------
template <int MODE>
__global__ __launch_bounds__(256) void gemm_bf16_wmma_kernel(
    const unsigned short* __restrict__ A, const unsigned short* __restrict__ Bt,
    void* __restrict__ Cout, int K, int ldA, int ldB, int ldC) {
  // LDS tiles: rows padded to 40 elements (80B pitch, matches TDM pad config)
  // -> conflict-free 16-lane row-parallel ds_load_b128, 16B aligned.
  __shared__ unsigned short Asm[2][64][40];    // 64  rows x 32 K  (+pad)
  __shared__ unsigned short Bsm[2][256][40];   // 256 rows x 32 K  (+pad)

  const int t    = threadIdx.x;
  const int lane = t & 31;
  const int wave = t >> 5;
  const int wm   = wave >> 2;                  // 0..1  (M slice, 32 rows)
  const int wn   = wave & 3;                   // 0..3  (N slice, 64 cols)
  const int moBlk = blockIdx.x * 64;
  const int noBlk = blockIdx.y * 256;
  const int l15  = lane & 15;
  const int koff = (lane & 16) ? 8 : 0;

  v8f acc[8];                                  // [am*4 + nb]
#pragma unroll
  for (int tt = 0; tt < 8; ++tt) acc[tt] = v8f{};

  const int nsteps = K / 32;

  // ---- prologue: TDM stages K-slice 0 into buffer 0 (wave 0 only) ----
  if (wave == 0) {
    tdm_load_2d((unsigned)(uintptr_t)&Asm[0][0][0],
                A + (size_t)moBlk * ldA, (unsigned)ldA, 32u, 64u);
    tdm_load_2d((unsigned)(uintptr_t)&Bsm[0][0][0],
                Bt + (size_t)noBlk * ldB, (unsigned)ldB, 32u, 256u);
  }

  for (int s = 0; s < nsteps; ++s) {
    const int buf = s & 1;
    if (wave == 0)
      __builtin_amdgcn_s_wait_tensorcnt(0);   // wave 0's DMA for `buf` landed
    __syncthreads();                          // publish LDS; prev compute done

    if (wave == 0 && s + 1 < nsteps) {        // stage next K-slice during compute
      const int k = (s + 1) * 32;
      tdm_load_2d((unsigned)(uintptr_t)&Asm[1 - buf][0][0],
                  A + (size_t)moBlk * ldA + k, (unsigned)ldA, 32u, 64u);
      tdm_load_2d((unsigned)(uintptr_t)&Bsm[1 - buf][0][0],
                  Bt + (size_t)noBlk * ldB + k, (unsigned)ldB, 32u, 256u);
    }

    // ---- compute on current buffer ----
    // Preload operands (2 A + 4 B = 12 ds_load_b128), then the 8-WMMA chain.
    bf16x16 a0 = make_op(&Asm[buf][wm * 32 +      l15][koff]);
    bf16x16 a1 = make_op(&Asm[buf][wm * 32 + 16 + l15][koff]);
    bf16x16 b[4];
#pragma unroll
    for (int j = 0; j < 4; ++j)
      b[j] = make_op(&Bsm[buf][wn * 64 + j * 16 + l15][koff]);

#pragma unroll
    for (int j = 0; j < 4; ++j) {
      acc[j]     = __builtin_amdgcn_wmma_f32_16x16x32_bf16(
          false, a0, false, b[j], (short)0, acc[j],     false, false);
      acc[4 + j] = __builtin_amdgcn_wmma_f32_16x16x32_bf16(
          false, a1, false, b[j], (short)0, acc[4 + j], false, false);
    }
  }

  // C/D layout (ISA 7.12.2): lane<16 -> N=lane,   VGPR r -> M = r
  //                          lane>=16-> N=lane-16, VGPR r -> M = r + 8
  const int half = lane >> 4;
#pragma unroll
  for (int tt = 0; tt < 8; ++tt) {
    const int am = tt >> 2, nb = tt & 3;
#pragma unroll
    for (int r = 0; r < 8; ++r) {
      int m = moBlk + wm * 32 + am * 16 + half * 8 + r;
      int n = noBlk + wn * 64 + nb * 16 + l15;
      float val = acc[tt][r];
      if constexpr (MODE == 0) {
        ((unsigned short*)Cout)[(size_t)m * ldC + n] = f2bf(val);
      } else if constexpr (MODE == 1) {
        ((unsigned short*)Cout)[(size_t)n * ldC + m] = f2bf(val);
      } else if constexpr (MODE == 2) {
        ((float*)Cout)[(size_t)m * ldC + n] = val / (1.0f + __expf(-val)); // silu
      } else {
        ((float*)Cout)[(size_t)m * ldC + n] = val;
      }
    }
  }
}

// ---------------------------------------------------------------------------
// Kernel 4: row softmax, f32 logits -> bf16 probs
// ---------------------------------------------------------------------------
__global__ __launch_bounds__(256) void softmax_rows_kernel(
    const float* __restrict__ L, unsigned short* __restrict__ P, int N) {
  int row = blockIdx.x;
  const float* lr = L + (size_t)row * N;
  int wave = threadIdx.x >> 5, lane = threadIdx.x & 31;
  __shared__ float red[8];

  float m = -1e30f;
  for (int i = threadIdx.x; i < N; i += 256) m = fmaxf(m, lr[i]);
  for (int o = 16; o > 0; o >>= 1) m = fmaxf(m, __shfl_xor(m, o, 32));
  if (lane == 0) red[wave] = m;
  __syncthreads();
  float rowmax = red[0];
#pragma unroll
  for (int i = 1; i < 8; ++i) rowmax = fmaxf(rowmax, red[i]);
  __syncthreads();

  float s = 0.f;
  for (int i = threadIdx.x; i < N; i += 256) s += __expf(lr[i] - rowmax);
  for (int o = 16; o > 0; o >>= 1) s += __shfl_xor(s, o, 32);
  if (lane == 0) red[wave] = s;
  __syncthreads();
  float tot = 0.f;
#pragma unroll
  for (int i = 0; i < 8; ++i) tot += red[i];
  float inv = 1.0f / tot;

  unsigned short* pr = P + (size_t)row * N;
  for (int i = threadIdx.x; i < N; i += 256)
    pr[i] = f2bf(__expf(lr[i] - rowmax) * inv);
}

// ---------------------------------------------------------------------------
// Host-side launcher.  Workspace layout (bytes):
//   hb   [S][D] bf16 :  8 MB @ 0
//   Wqb  [D][D] bf16 :  2 MB @ 8M      Wkb @ 10M      Wvb @ 12M
//   Qb   [S][D] bf16 :  8 MB @ 14M     Kb  @ 22M
//   Vtb  [D][S] bf16 :  8 MB @ 30M     (V stored transposed)
//   L    [S][S] f32  : 64 MB @ 38M
//   P    [S][S] bf16 : 32 MB @ 102M    -> total 134 MB (fits in 192MB L2)
// ---------------------------------------------------------------------------
extern "C" void kernel_launch(void* const* d_in, const int* in_sizes, int n_in,
                              void* d_out, int out_size, void* d_ws, size_t ws_size,
                              hipStream_t stream) {
  const int*   x    = (const int*)d_in[0];
  const float* emb  = (const float*)d_in[1];
  const float* nw   = (const float*)d_in[2];
  const float* Wq   = (const float*)d_in[3];
  const float* Wk   = (const float*)d_in[4];
  const float* Wv   = (const float*)d_in[5];
  float*       out  = (float*)d_out;

  char* ws = (char*)d_ws;
  const size_t MB = 1024ull * 1024ull;
  unsigned short* hb  = (unsigned short*)(ws + 0);
  unsigned short* Wqb = (unsigned short*)(ws + 8 * MB);
  unsigned short* Wkb = (unsigned short*)(ws + 10 * MB);
  unsigned short* Wvb = (unsigned short*)(ws + 12 * MB);
  unsigned short* Qb  = (unsigned short*)(ws + 14 * MB);
  unsigned short* Kb  = (unsigned short*)(ws + 22 * MB);
  unsigned short* Vtb = (unsigned short*)(ws + 30 * MB);
  float*          L   = (float*)(ws + 38 * MB);
  unsigned short* P   = (unsigned short*)(ws + 102 * MB);

  // 1) embedding + RMSNorm -> bf16 h
  embed_rmsnorm_kernel<<<SEQ, 256, 0, stream>>>(x, emb, nw, hb);

  // 2) weights -> bf16
  {
    int n4 = DIM * DIM / 4;
    int blocks = (n4 + 255) / 256;
    cvt_f32_bf16_kernel<<<blocks, 256, 0, stream>>>(Wq, Wqb, n4);
    cvt_f32_bf16_kernel<<<blocks, 256, 0, stream>>>(Wk, Wkb, n4);
    cvt_f32_bf16_kernel<<<blocks, 256, 0, stream>>>(Wv, Wvb, n4);
  }

  // 3) Q = h @ Wq^T ; K = h @ Wk^T ; V^T = (h @ Wv^T)^T    [M=S, N=D, K=D]
  dim3 gQKV(SEQ / 64, DIM / 256);
  gemm_bf16_wmma_kernel<0><<<gQKV, 256, 0, stream>>>(hb, Wqb, Qb,  DIM, DIM, DIM, DIM);
  gemm_bf16_wmma_kernel<0><<<gQKV, 256, 0, stream>>>(hb, Wkb, Kb,  DIM, DIM, DIM, DIM);
  gemm_bf16_wmma_kernel<1><<<gQKV, 256, 0, stream>>>(hb, Wvb, Vtb, DIM, DIM, DIM, SEQ);

  // 4) L = Q @ K^T                                         [M=S, N=S, K=D]
  dim3 gAtt(SEQ / 64, SEQ / 256);
  gemm_bf16_wmma_kernel<3><<<gAtt, 256, 0, stream>>>(Qb, Kb, L, DIM, DIM, DIM, SEQ);

  // 5) P = softmax_rows(L)  (bf16)
  softmax_rows_kernel<<<SEQ, 256, 0, stream>>>(L, P, SEQ);

  // 6) out = silu(P @ V)                                   [M=S, N=D, K=S]
  dim3 gOut(SEQ / 64, DIM / 256);
  gemm_bf16_wmma_kernel<2><<<gOut, 256, 0, stream>>>(P, Vtb, out, SEQ, SEQ, SEQ, DIM);
}
